// NerDetector_51573967290808
// MI455X (gfx1250) — matrix operand order repaired and checked by reading, not compiled, not run on hardware
//
#include <hip/hip_runtime.h>
#include <hip/hip_bf16.h>

typedef __attribute__((ext_vector_type(16))) _Float16 v16h;
typedef __attribute__((ext_vector_type(8)))  _Float16 v8h;
typedef __attribute__((ext_vector_type(8)))  float    v8f;

#define BB   64
#define TT   512
#define DD   768
#define WW   255
#define HH   8
#define DHH  96
#define FFD  3072
#define MTOT (BB*WW)   /* 16320, divisible by 32 */
#define PAD  256

// ---------------------------------------------------------------------------
// Fragment loaders (CDNA5 WMMA 16x16x32 f16 layouts, wave32)
// A (16x32, MxK): lanes 0-15 -> M=lane, K chunks {bk..bk+7, 16+bk..16+bk+7},
//                 bk = (lane>>4)*8
// B^T layout:     lane gives N=lane&15, K chunks {bk..bk+7, 16+bk..16+bk+7}, bk=(lane>>4)*8
// C/D (16x16 f32): lane n = lane&15, vgpr j -> M = j + 8*(lane>>4)
// ---------------------------------------------------------------------------
__device__ __forceinline__ v16h load_a_frag(const _Float16* __restrict__ Abase,
                                            int lda, int lane) {
  int m  = lane & 15;
  int bk = (lane >> 4) * 8;
  const v8h* p0 = reinterpret_cast<const v8h*>(Abase + (size_t)m * lda + bk);
  const v8h* p1 = reinterpret_cast<const v8h*>(Abase + (size_t)m * lda + 16 + bk);
  v8h lo = *p0, hi = *p1;
  v16h a;
#pragma unroll
  for (int i = 0; i < 8; ++i) { a[i] = lo[i]; a[8 + i] = hi[i]; }
  return a;
}

__device__ __forceinline__ v16h load_b_frag_t(const _Float16* __restrict__ Bbase,
                                              int ldb, int lane) {
  int n    = lane & 15;
  int bk16 = (lane >> 4) * 8;
  const v8h* p0 = reinterpret_cast<const v8h*>(Bbase + (size_t)n * ldb + bk16);
  const v8h* p1 = reinterpret_cast<const v8h*>(Bbase + (size_t)n * ldb + 16 + bk16);
  v8h lo = *p0, hi = *p1;
  v16h b;
#pragma unroll
  for (int i = 0; i < 8; ++i) { b[i] = lo[i]; b[8 + i] = hi[i]; }
  return b;
}

// ---------------------------------------------------------------------------
// Generic GEMM with 2x2 register blocking:
// each wave computes a 32x32 tile of C[M,N] = act(A@W^T + bias + residual)
// (4 WMMAs per K-step, A/B fragments each reused twice)
// M % 32 == 0, N % 32 == 0, K % 32 == 0
// ---------------------------------------------------------------------------
__global__ __launch_bounds__(256)
void gemm_wmma(const _Float16* __restrict__ A, int lda,
               const _Float16* __restrict__ WT, int ldw,
               const float* __restrict__ bias,
               const float* __restrict__ residual,
               float* __restrict__ C32, _Float16* __restrict__ C16,
               int M, int N, int K, int relu)
{
  int wid  = threadIdx.x >> 5;
  int lane = threadIdx.x & 31;
  int nblk = N >> 5;                       // 32-wide column blocks
  int tile = blockIdx.x * 8 + wid;
  int total = (M >> 5) * nblk;
  if (tile >= total) return;
  int mt = tile / nblk, nt = tile % nblk;  // 32x32 macro-tile
  const _Float16* A0 = A  + (size_t)(mt * 32) * lda;
  const _Float16* A1 = A0 + (size_t)16 * lda;
  const _Float16* B0 = WT + (size_t)(nt * 32) * ldw;
  const _Float16* B1 = B0 + (size_t)16 * ldw;
  v8f c00 = {}, c01 = {}, c10 = {}, c11 = {};
  for (int k0 = 0; k0 < K; k0 += 32) {
    v16h a0 = load_a_frag(A0 + k0, lda, lane);
    v16h a1 = load_a_frag(A1 + k0, lda, lane);
    v16h b0 = load_b_frag_t(B0 + k0, ldw, lane);
    v16h b1 = load_b_frag_t(B1 + k0, ldw, lane);
    c00 = __builtin_amdgcn_wmma_f32_16x16x32_f16(false, a0, false, b0, (short)0, c00, false, false);
    c01 = __builtin_amdgcn_wmma_f32_16x16x32_f16(false, a0, false, b1, (short)0, c01, false, false);
    c10 = __builtin_amdgcn_wmma_f32_16x16x32_f16(false, a1, false, b0, (short)0, c10, false, false);
    c11 = __builtin_amdgcn_wmma_f32_16x16x32_f16(false, a1, false, b1, (short)0, c11, false, false);
  }
  int n     = lane & 15;
  int mbase = (lane >> 4) * 8;
  v8f acc[2][2] = {{c00, c01}, {c10, c11}};
#pragma unroll
  for (int bm = 0; bm < 2; ++bm) {
#pragma unroll
    for (int bn = 0; bn < 2; ++bn) {
      int ncol = nt * 32 + bn * 16 + n;
      float bv = bias ? bias[ncol] : 0.f;
#pragma unroll
      for (int j = 0; j < 8; ++j) {
        int row = mt * 32 + bm * 16 + mbase + j;
        float v = acc[bm][bn][j] + bv;
        if (residual) v += residual[(size_t)row * N + ncol];
        if (relu) v = v > 0.f ? v : 0.f;
        if (C32) C32[(size_t)row * N + ncol] = v;
        if (C16) C16[(size_t)row * N + ncol] = (_Float16)v;
      }
    }
  }
}

// ---------------------------------------------------------------------------
// Attention scores: per (b,h): S[256,256] = Q_h[256,96] @ K_h^T   (unscaled)
// ---------------------------------------------------------------------------
__global__ __launch_bounds__(256)
void attn_scores_kernel(const _Float16* __restrict__ qH,
                        const _Float16* __restrict__ kH,
                        float* __restrict__ scores)
{
  int wid = threadIdx.x >> 5, lane = threadIdx.x & 31;
  int t = blockIdx.x * 8 + wid;
  if (t >= BB * HH * 16 * 16) return;
  int bh = t >> 8, r = t & 255;
  int qt = r >> 4, kt = r & 15;
  int b = bh / HH, h = bh % HH;
  int m = lane & 15, bk = (lane >> 4) * 8;
  int n = lane & 15, bk16 = (lane >> 4) * 16;
  int qrow = qt * 16 + m; if (qrow > WW - 1) qrow = WW - 1;
  int krow = kt * 16 + n; if (krow > WW - 1) krow = WW - 1;
  const _Float16* Arow = qH + ((size_t)(b * WW + qrow)) * DD + h * DHH;
  const _Float16* Brow = kH + ((size_t)(b * WW + krow)) * DD + h * DHH;
  v8f c = {};
#pragma unroll
  for (int k0 = 0; k0 < DHH; k0 += 32) {
    const v8h* a0 = (const v8h*)(Arow + k0 + bk);
    const v8h* a1 = (const v8h*)(Arow + k0 + 16 + bk);
    const v8h* b0 = (const v8h*)(Brow + k0 + bk);
    const v8h* b1 = (const v8h*)(Brow + k0 + 16 + bk);
    v8h alo = *a0, ahi = *a1, blo = *b0, bhi = *b1;
    v16h a, bb;
#pragma unroll
    for (int i = 0; i < 8; ++i) {
      a[i] = alo[i]; a[8 + i] = ahi[i];
      bb[i] = blo[i]; bb[8 + i] = bhi[i];
    }
    c = __builtin_amdgcn_wmma_f32_16x16x32_f16(false, a, false, bb,
                                               (short)0, c, false, false);
  }
  int mb = (lane >> 4) * 8;
  float* out = scores + ((size_t)bh * PAD + qt * 16) * PAD + kt * 16;
#pragma unroll
  for (int j = 0; j < 8; ++j) out[(size_t)(mb + j) * PAD + n] = c[j];
}

// ---------------------------------------------------------------------------
// Softmax over 256-wide rows (mask col >= W), scale 1/sqrt(96), emit f16 attn
// ---------------------------------------------------------------------------
__global__ __launch_bounds__(256)
void softmax_kernel(const float* __restrict__ scores, _Float16* __restrict__ attnH)
{
  __shared__ float red[256];
  size_t row = blockIdx.x;          // B*H*256 rows
  int col = threadIdx.x;
  const float scale = 0.1020620726159658f; // 1/sqrt(96)
  float s = scores[row * PAD + col] * scale;
  if (col >= WW) s = -1e30f;
  red[col] = s; __syncthreads();
  for (int st = 128; st > 0; st >>= 1) {
    if (col < st) red[col] = fmaxf(red[col], red[col + st]);
    __syncthreads();
  }
  float mx = red[0]; __syncthreads();
  float e = (col >= WW) ? 0.f : __expf(s - mx);
  red[col] = e; __syncthreads();
  for (int st = 128; st > 0; st >>= 1) {
    if (col < st) red[col] += red[col + st];
    __syncthreads();
  }
  float sum = red[0];
  attnH[row * PAD + col] = (_Float16)(e / sum);
}

// ---------------------------------------------------------------------------
// Context: per (b,h): ctx[256,96] = attn[256,256] @ V_h[256,96]
// ---------------------------------------------------------------------------
__global__ __launch_bounds__(256)
void attn_ctx_kernel(const _Float16* __restrict__ attnH,
                     const _Float16* __restrict__ vH,
                     _Float16* __restrict__ ctxH)
{
  int wid = threadIdx.x >> 5, lane = threadIdx.x & 31;
  int t = blockIdx.x * 8 + wid;
  if (t >= BB * HH * 16 * 6) return;
  int bh = t / 96, r = t % 96;
  int qt = r / 6, nt = r % 6;
  int b = bh / HH, h = bh % HH;
  int m = lane & 15, bk = (lane >> 4) * 8;
  int n = lane & 15, bk16 = (lane >> 4) * 16;
  const _Float16* Abase = attnH + (size_t)bh * PAD * PAD + (size_t)(qt * 16) * PAD;
  v8f c = {};
  for (int k0 = 0; k0 < PAD; k0 += 32) {
    const v8h* a0 = (const v8h*)(Abase + (size_t)m * PAD + k0 + bk);
    const v8h* a1 = (const v8h*)(Abase + (size_t)m * PAD + k0 + 16 + bk);
    v8h alo = *a0, ahi = *a1;
    v16h a, bb;
#pragma unroll
    for (int i = 0; i < 8; ++i) { a[i] = alo[i]; a[8 + i] = ahi[i]; }
#pragma unroll
    for (int i = 0; i < 16; ++i) {
      int kk = k0 + bk + i + (i >= 8 ? 8 : 0);
      int vr = kk < WW ? kk : WW - 1;  // attn[.,255]==0 so value is irrelevant
      bb[i] = vH[((size_t)(b * WW + vr)) * DD + h * DHH + nt * 16 + n];
    }
    c = __builtin_amdgcn_wmma_f32_16x16x32_f16(false, a, false, bb,
                                               (short)0, c, false, false);
  }
  int mb = (lane >> 4) * 8;
#pragma unroll
  for (int j = 0; j < 8; ++j) {
    int q = qt * 16 + mb + j;
    if (q < WW)
      ctxH[((size_t)(b * WW + q)) * DD + h * DHH + nt * 16 + n] = (_Float16)c[j];
  }
}

// ---------------------------------------------------------------------------
// First-subtoken gather: feat[b,w,:] = bert[b, argmax_t(wid[b,t]==w), :]
// ---------------------------------------------------------------------------
__global__ __launch_bounds__(256)
void gather_kernel(const float* __restrict__ bert, const int* __restrict__ wids,
                   float* __restrict__ feat32, _Float16* __restrict__ featH)
{
  __shared__ int s_idx;
  int bw = blockIdx.x;
  int b = bw / WW, w = bw % WW;
  if (threadIdx.x == 0) s_idx = TT;
  __syncthreads();
  for (int tpos = threadIdx.x; tpos < TT; tpos += 256)
    if (wids[b * TT + tpos] == w) atomicMin(&s_idx, tpos);
  __syncthreads();
  int idx = (s_idx >= TT) ? 0 : s_idx;   // argmax of all-false -> 0
  const float* src = bert + ((size_t)b * TT + idx) * DD;
  size_t row = (size_t)bw * DD;
  for (int i = threadIdx.x; i < DD; i += 256) {
    float v = src[i];
    feat32[row + i] = v;
    featH[row + i] = (_Float16)v;
  }
}

// ---------------------------------------------------------------------------
// LayerNorm over D=768 (3 elems/thread), writes f32 and optional f16
// ---------------------------------------------------------------------------
__global__ __launch_bounds__(256)
void ln_kernel(const float* __restrict__ X, const float* __restrict__ g,
               const float* __restrict__ be, float* __restrict__ Y32,
               _Float16* __restrict__ Y16)
{
  __shared__ float red[256];
  size_t row = (size_t)blockIdx.x * DD;
  int tid = threadIdx.x;
  float x0 = X[row + tid], x1 = X[row + tid + 256], x2 = X[row + tid + 512];
  red[tid] = x0 + x1 + x2; __syncthreads();
  for (int st = 128; st > 0; st >>= 1) { if (tid < st) red[tid] += red[tid + st]; __syncthreads(); }
  float mean = red[0] * (1.0f / DD); __syncthreads();
  float d0 = x0 - mean, d1 = x1 - mean, d2 = x2 - mean;
  red[tid] = d0 * d0 + d1 * d1 + d2 * d2; __syncthreads();
  for (int st = 128; st > 0; st >>= 1) { if (tid < st) red[tid] += red[tid + st]; __syncthreads(); }
  float rstd = rsqrtf(red[0] * (1.0f / DD) + 1e-5f);
  float y0 = d0 * rstd * g[tid] + be[tid];
  float y1 = d1 * rstd * g[tid + 256] + be[tid + 256];
  float y2 = d2 * rstd * g[tid + 512] + be[tid + 512];
  Y32[row + tid] = y0; Y32[row + tid + 256] = y1; Y32[row + tid + 512] = y2;
  if (Y16) {
    Y16[row + tid] = (_Float16)y0;
    Y16[row + tid + 256] = (_Float16)y1;
    Y16[row + tid + 512] = (_Float16)y2;
  }
}

// ---------------------------------------------------------------------------
// Convert f32 [K,N] weight -> f16 transposed [N,K]
// ---------------------------------------------------------------------------
__global__ __launch_bounds__(256)
void convT_kernel(const float* __restrict__ Win, _Float16* __restrict__ Wout,
                  int K, int N)
{
  int i = blockIdx.x * 256 + threadIdx.x;
  if (i >= K * N) return;
  int k = i / N, n = i % N;
  Wout[(size_t)n * K + k] = (_Float16)Win[i];
}

// ---------------------------------------------------------------------------
// Final LN(ng,nb) + Linear(768->2) + softmax + argmax
// out: [MTOT*2] probs then [MTOT] path (as float)
// ---------------------------------------------------------------------------
__global__ __launch_bounds__(256)
void final_kernel(const float* __restrict__ X, const float* __restrict__ ng,
                  const float* __restrict__ nb, const float* __restrict__ lw,
                  const float* __restrict__ lb, float* __restrict__ out)
{
  __shared__ float red[256];
  __shared__ float red2[256];
  size_t row = (size_t)blockIdx.x * DD;
  int tid = threadIdx.x;
  float x0 = X[row + tid], x1 = X[row + tid + 256], x2 = X[row + tid + 512];
  red[tid] = x0 + x1 + x2; __syncthreads();
  for (int st = 128; st > 0; st >>= 1) { if (tid < st) red[tid] += red[tid + st]; __syncthreads(); }
  float mean = red[0] * (1.0f / DD); __syncthreads();
  float d0 = x0 - mean, d1 = x1 - mean, d2 = x2 - mean;
  red[tid] = d0 * d0 + d1 * d1 + d2 * d2; __syncthreads();
  for (int st = 128; st > 0; st >>= 1) { if (tid < st) red[tid] += red[tid + st]; __syncthreads(); }
  float rstd = rsqrtf(red[0] * (1.0f / DD) + 1e-5f); __syncthreads();
  float z0 = d0 * rstd * ng[tid] + nb[tid];
  float z1 = d1 * rstd * ng[tid + 256] + nb[tid + 256];
  float z2 = d2 * rstd * ng[tid + 512] + nb[tid + 512];
  float p0 = z0 * lw[tid * 2]     + z1 * lw[(tid + 256) * 2]     + z2 * lw[(tid + 512) * 2];
  float p1 = z0 * lw[tid * 2 + 1] + z1 * lw[(tid + 256) * 2 + 1] + z2 * lw[(tid + 512) * 2 + 1];
  red[tid] = p0; red2[tid] = p1; __syncthreads();
  for (int st = 128; st > 0; st >>= 1) {
    if (tid < st) { red[tid] += red[tid + st]; red2[tid] += red2[tid + st]; }
    __syncthreads();
  }
  if (tid == 0) {
    float l0 = red[0] + lb[0], l1 = red2[0] + lb[1];
    float mx = fmaxf(l0, l1);
    float e0 = __expf(l0 - mx), e1 = __expf(l1 - mx);
    float s = e0 + e1;
    out[(size_t)blockIdx.x * 2]     = e0 / s;
    out[(size_t)blockIdx.x * 2 + 1] = e1 / s;
    out[(size_t)MTOT * 2 + blockIdx.x] = (e1 > e0) ? 1.0f : 0.0f;
  }
}

// ---------------------------------------------------------------------------
extern "C" void kernel_launch(void* const* d_in, const int* in_sizes, int n_in,
                              void* d_out, int out_size, void* d_ws, size_t ws_size,
                              hipStream_t stream) {
  (void)in_sizes; (void)n_in; (void)out_size; (void)ws_size;
  const float* bert  = (const float*)d_in[0];
  const int*   wids  = (const int*)d_in[1];
  const float* Wq    = (const float*)d_in[2];  const float* bq  = (const float*)d_in[3];
  const float* Wk    = (const float*)d_in[4];  const float* bk  = (const float*)d_in[5];
  const float* Wv    = (const float*)d_in[6];  const float* bv  = (const float*)d_in[7];
  const float* Wo    = (const float*)d_in[8];  const float* bo  = (const float*)d_in[9];
  const float* ln1g  = (const float*)d_in[10]; const float* ln1b = (const float*)d_in[11];
  const float* W1f   = (const float*)d_in[12]; const float* b1f = (const float*)d_in[13];
  const float* W2f   = (const float*)d_in[14]; const float* b2f = (const float*)d_in[15];
  const float* ln2g  = (const float*)d_in[16]; const float* ln2b = (const float*)d_in[17];
  const float* ngv   = (const float*)d_in[18]; const float* nbv = (const float*)d_in[19];
  const float* linw  = (const float*)d_in[20]; const float* linb = (const float*)d_in[21];
  float* out = (float*)d_out;

  // workspace bump allocator (256B aligned)
  char* ws = (char*)d_ws;
  size_t off = 0;
  auto alloc = [&](size_t bytes) -> char* {
    char* p = ws + off;
    off += (bytes + 255) & ~(size_t)255;
    return p;
  };
  const size_t MD4 = (size_t)MTOT * DD * 4, MD2 = (size_t)MTOT * DD * 2;
  float*     feat32 = (float*)alloc(MD4);
  _Float16*  featH  = (_Float16*)alloc(MD2);
  _Float16*  qH     = (_Float16*)alloc(MD2);
  _Float16*  kHbuf  = (_Float16*)alloc(MD2);
  _Float16*  vHbuf  = (_Float16*)alloc(MD2);
  _Float16*  ctxH   = (_Float16*)alloc(MD2);
  float*     scores = (float*)alloc((size_t)BB * HH * PAD * PAD * 4);
  _Float16*  attnH  = (_Float16*)alloc((size_t)BB * HH * PAD * PAD * 2);
  float*     tmp32  = (float*)alloc(MD4);          // pre-LN1 result, reused pre-LN2
  float*     x32    = (float*)alloc(MD4);          // after LN1
  _Float16*  xH     = (_Float16*)alloc(MD2);
  _Float16*  h1H    = (_Float16*)alloc((size_t)MTOT * FFD * 2);
  float*     x2_32  = (float*)alloc(MD4);          // after LN2
  _Float16*  WqT    = (_Float16*)alloc((size_t)DD * DD * 2);
  _Float16*  WkT    = (_Float16*)alloc((size_t)DD * DD * 2);
  _Float16*  WvT    = (_Float16*)alloc((size_t)DD * DD * 2);
  _Float16*  WoT    = (_Float16*)alloc((size_t)DD * DD * 2);
  _Float16*  W1fT   = (_Float16*)alloc((size_t)DD * FFD * 2);
  _Float16*  W2fT   = (_Float16*)alloc((size_t)FFD * DD * 2);

  // 1) weight convert + transpose to f16 [N,K]
  int gDD = (DD * DD + 255) / 256, gDF = (DD * FFD + 255) / 256;
  convT_kernel<<<gDD, 256, 0, stream>>>(Wq,  WqT,  DD, DD);
  convT_kernel<<<gDD, 256, 0, stream>>>(Wk,  WkT,  DD, DD);
  convT_kernel<<<gDD, 256, 0, stream>>>(Wv,  WvT,  DD, DD);
  convT_kernel<<<gDD, 256, 0, stream>>>(Wo,  WoT,  DD, DD);
  convT_kernel<<<gDF, 256, 0, stream>>>(W1f, W1fT, DD, FFD);
  convT_kernel<<<gDF, 256, 0, stream>>>(W2f, W2fT, FFD, DD);

  // 2) first-subtoken gather
  gather_kernel<<<MTOT, 256, 0, stream>>>(bert, wids, feat32, featH);

  // 3) Q, K, V projections (f16 WMMA, 32x32 macro-tiles per wave)
  int tilesDD = (MTOT / 32) * (DD / 32);          // 12240
  int gGemmDD = (tilesDD + 7) / 8;                // 1530
  gemm_wmma<<<gGemmDD, 256, 0, stream>>>(featH, DD, WqT, DD, bq, nullptr,
                                         nullptr, qH,    MTOT, DD, DD, 0);
  gemm_wmma<<<gGemmDD, 256, 0, stream>>>(featH, DD, WkT, DD, bk, nullptr,
                                         nullptr, kHbuf, MTOT, DD, DD, 0);
  gemm_wmma<<<gGemmDD, 256, 0, stream>>>(featH, DD, WvT, DD, bv, nullptr,
                                         nullptr, vHbuf, MTOT, DD, DD, 0);

  // 4) attention scores, softmax, context
  attn_scores_kernel<<<(BB * HH * 256 + 7) / 8, 256, 0, stream>>>(qH, kHbuf, scores);
  softmax_kernel<<<BB * HH * PAD, 256, 0, stream>>>(scores, attnH);
  attn_ctx_kernel<<<(BB * HH * 96 + 7) / 8, 256, 0, stream>>>(attnH, vHbuf, ctxH);

  // 5) output projection + residual, LN1
  gemm_wmma<<<gGemmDD, 256, 0, stream>>>(ctxH, DD, WoT, DD, bo, feat32,
                                         tmp32, nullptr, MTOT, DD, DD, 0);
  ln_kernel<<<MTOT, 256, 0, stream>>>(tmp32, ln1g, ln1b, x32, xH);

  // 6) FFN: relu(x @ W1f + b1f) @ W2f + b2f + x, LN2
  int tilesDF = (MTOT / 32) * (FFD / 32);         // 48960
  gemm_wmma<<<(tilesDF + 7) / 8, 256, 0, stream>>>(xH, DD, W1fT, DD, b1f, nullptr,
                                                   nullptr, h1H, MTOT, FFD, DD, 1);
  gemm_wmma<<<gGemmDD, 256, 0, stream>>>(h1H, FFD, W2fT, FFD, b2f, x32,
                                         tmp32, nullptr, MTOT, DD, FFD, 0);
  ln_kernel<<<MTOT, 256, 0, stream>>>(tmp32, ln2g, ln2b, x2_32, nullptr);

  // 7) final LN + linear + softmax + argmax
  final_kernel<<<MTOT, 256, 0, stream>>>(x2_32, ngv, nbv, linw, linb, out);
}
